// XENet_23965917511872
// MI455X (gfx1250) — matrix operand order repaired and checked by reading, not compiled, not run on hardware
//
#include <hip/hip_runtime.h>
#include <stdint.h>

// ---------------- problem constants ----------------
#define N_NODES 32768
#define FIN     64
#define SIN     32
#define S0      128
#define S1      128
#define FOUT    64
#define SOUT    32
#define E_HALF  262144
#define NE      524288       // 2*E_HALF
#define DIN     192          // 2*FIN + 2*SIN

// packed-weight sizes (halves). fragment = 32x16 tile = 512 halves, lane-major.
#define W0_FRAGS (6*8)
#define W1_FRAGS (4*8)
#define WE_FRAGS (4*2)
#define WX_FRAGS (10*4)
#define W0_H (W0_FRAGS*512)  // 24576
#define W1_H (W1_FRAGS*512)  // 16384
#define WE_H (WE_FRAGS*512)  // 4096
#define WX_H (WX_FRAGS*512)  // 20480
#define PACK_TOTAL_H (W0_H + W1_H + WE_H + WX_H)   // 65536
#define WEDGE_H (W0_H + W1_H + WE_H)               // 45056 halves = 90112 B

typedef __attribute__((ext_vector_type(16))) _Float16 v16h;
typedef __attribute__((ext_vector_type(8)))  _Float16 v8h;
typedef __attribute__((ext_vector_type(8)))  float    v8f;
typedef __attribute__((ext_vector_type(4)))  float    v4f;
typedef unsigned int u32x4 __attribute__((ext_vector_type(4)));
typedef int          i32x4 __attribute__((ext_vector_type(4)));
typedef int          i32x8 __attribute__((ext_vector_type(8)));
typedef int          gv4i  __attribute__((vector_size(16)));   // GCC-style int4

// D = A(16x32 f16) * B(32x16 f16) + C(16x16 f32)
__device__ __forceinline__ v8f wmma_f16(v16h a, v16h b, v8f c) {
    return __builtin_amdgcn_wmma_f32_16x16x32_f16(false, a, false, b, (short)0, c,
                                                  false, false);
}

// A-fragment from a row-major f16 LDS matrix (rowStride in halves).
__device__ __forceinline__ v16h ldsA(const _Float16* base, int rowStride, int kt, int lane) {
    const _Float16* p = base + (lane & 15) * rowStride + kt * 32 + ((lane >> 4) << 3);
    v16h a;
    *(v8h*)&a       = *(const v8h*)(p);
    *((v8h*)&a + 1) = *(const v8h*)(p + 16);
    return a;
}

// A-fragment from a row-major f32 LDS matrix: two 32B runs + convert to f16.
__device__ __forceinline__ v16h ldsA32(const float* base, int rowStride, int kt, int lane) {
    const float* p = base + (lane & 15) * rowStride + kt * 32 + ((lane >> 4) << 3);
    v4f f0 = *(const v4f*)(p);
    v4f f1 = *(const v4f*)(p + 4);
    v4f f2 = *(const v4f*)(p + 16);
    v4f f3 = *(const v4f*)(p + 20);
    v16h a;
    #pragma unroll
    for (int i = 0; i < 4; i++) {
        a[i]      = (_Float16)f0[i];
        a[4 + i]  = (_Float16)f1[i];
        a[8 + i]  = (_Float16)f2[i];
        a[12 + i] = (_Float16)f3[i];
    }
    return a;
}

// B-fragment load from pre-packed LDS: each lane's 16 halves stored contiguously.
__device__ __forceinline__ v16h ldsB(const _Float16* fragBase, int lane) {
    const _Float16* p = fragBase + lane * 16;
    v16h b;
    *(v8h*)&b       = *(const v8h*)(p);
    *((v8h*)&b + 1) = *(const v8h*)(p + 8);
    return b;
}

// ---------------- CDNA5 async global->LDS gather ----------------
#if __has_builtin(__builtin_amdgcn_global_load_async_to_lds_b128) && \
    __has_builtin(__builtin_amdgcn_s_wait_asynccnt)
#define HAVE_ASYNC 1
#warning "CDNA5: global_load_async_to_lds_b128 path ENABLED"
#else
#warning "CDNA5: async-to-LDS builtin NOT available; plain copy fallback"
#endif

// 16-byte per-lane gather into LDS (ASYNCcnt-tracked when available).
__device__ __forceinline__ void copy16ToLds(float* ldsDst, const float* gsrc) {
#ifdef HAVE_ASYNC
    __builtin_amdgcn_global_load_async_to_lds_b128(
        (__attribute__((address_space(1))) gv4i*)(uintptr_t)gsrc,
        (__attribute__((address_space(3))) gv4i*)(unsigned int)(uintptr_t)ldsDst,
        0, 0);
#else
    *(v4f*)ldsDst = *(const v4f*)gsrc;
#endif
}
__device__ __forceinline__ void stageWait() {
#ifdef HAVE_ASYNC
    __builtin_amdgcn_s_wait_asynccnt(0);
#endif
}

// ---------------- Tensor Data Mover weight preload ----------------
#if __has_builtin(__builtin_amdgcn_tensor_load_to_lds) && \
    __has_builtin(__builtin_amdgcn_s_wait_tensorcnt)
#define HAVE_TDM 1
#if defined(__clang_major__) && (__clang_major__ >= 23)
#warning "CDNA5 TDM: tensor_load_to_lds ENABLED (6-arg form)"
#else
#warning "CDNA5 TDM: tensor_load_to_lds ENABLED (5-arg form)"
#endif
// D# per CDNA5 ISA ch.8: group0 = count/lds_addr/global_addr/type,
// group1 = data_size + tensor dims/strides + tile dims. 2D tile == tensor.
__device__ __forceinline__ void tdm_load_2d(unsigned int lds_byte_addr,
                                            const void* gsrc,
                                            unsigned int rowElems,
                                            unsigned int rows) {
    unsigned long long ga = (unsigned long long)(uintptr_t)gsrc;
    u32x4 g0 = {};
    g0[0] = 1u;                                        // count=1 (valid user D#)
    g0[1] = lds_byte_addr;                             // lds_addr
    g0[2] = (unsigned int)ga;                          // global_addr[31:0]
    g0[3] = (unsigned int)(ga >> 32) | (2u << 30);     // global_addr[56:32] | type=2
    i32x8 g1 = {};
    g1[0] = (int)(2u << 16);                           // data_size = 4 bytes
    g1[1] = (int)((rowElems & 0xFFFFu) << 16);         // tensor_dim0[15:0]
    g1[2] = (int)((rowElems >> 16) | ((rows & 0xFFFFu) << 16));   // dim0 hi | dim1 lo
    g1[3] = (int)((rows >> 16) | ((rowElems & 0xFFFFu) << 16));   // dim1 hi | tile_dim0
    g1[4] = (int)(rows & 0xFFFFu);                     // tile_dim1 (tile_dim2 = 0)
    g1[5] = (int)rowElems;                             // tensor_dim0_stride[31:0]
    i32x4 z4 = {};
#if defined(__clang_major__) && (__clang_major__ >= 23)
    i32x8 z8 = {};
    __builtin_amdgcn_tensor_load_to_lds(g0, g1, z4, z4, z8, 0);
#else
    __builtin_amdgcn_tensor_load_to_lds(g0, g1, z4, z4, 0);
#endif
    __builtin_amdgcn_s_wait_tensorcnt(0);
}
#else
#warning "CDNA5 TDM: builtin NOT available; plain copy fallback"
#endif

// ---------------- kernel 0a: zero scratch accumulators + flag ----------------
__global__ void zero_kernel(float4* buf, int n4, int* flagInt) {
    int i = blockIdx.x * blockDim.x + threadIdx.x;
    if (i == 0) *flagInt = 0;
    for (; i < n4; i += gridDim.x * blockDim.x) buf[i] = make_float4(0.f, 0.f, 0.f, 0.f);
}

// ---------------- kernel 0b: pack all weight matrices into B-fragment order ----------------
__global__ void pack_kernel(const float* W0, const float* W1, const float* We,
                            const float* Wx, _Float16* out) {
    int t = blockIdx.x * blockDim.x + threadIdx.x;
    if (t >= PACK_TOTAL_H) return;
    const float* W; int nK, Ntot, u;
    if (t < W0_H)                    { W = W0; nK = 6;  Ntot = 128; u = t; }
    else if (t < W0_H + W1_H)        { W = W1; nK = 4;  Ntot = 128; u = t - W0_H; }
    else if (t < W0_H + W1_H + WE_H) { W = We; nK = 4;  Ntot = 32;  u = t - W0_H - W1_H; }
    else                             { W = Wx; nK = 10; Ntot = 64;  u = t - W0_H - W1_H - WE_H; }
    int frag = u >> 9, idx = u & 511;
    int lane = idx >> 4, i = idx & 15;
    int nt = frag / nK, kt = frag % nK;
    int K = kt * 32 + ((lane >> 4) << 3) + (i & 7) + ((i >> 3) << 4);
    int col = nt * 16 + (lane & 15);
    out[t] = (_Float16)W[(size_t)K * Ntot + col];
}

// ---------------- kernel 0c: verify reverse-edge structure ----------------
__global__ void check_kernel(const int* a, int* flagInt) {
    int i = blockIdx.x * blockDim.x + threadIdx.x;
    if (i >= NE) return;
    int s = a[2 * i], d = a[2 * i + 1];
    int rev = (i >= E_HALF) ? (i - E_HALF) : (i + E_HALF);
    if (a[2 * rev] != d || a[2 * rev + 1] != s) atomicOr(flagInt, 1);
}

// ---------------- kernel 1: edge MLP (3 chained WMMA GEMMs) + scatter ----------------
__global__ __launch_bounds__(128) void edge_kernel(
    const float* __restrict__ x_in, const int* __restrict__ a_in,
    const float* __restrict__ e_in,
    const float* __restrict__ b0, const float* __restrict__ b1,
    const float* __restrict__ prelu_slope, const float* __restrict__ be,
    const _Float16* __restrict__ packW,
    float* __restrict__ incoming, float* __restrict__ outgoing,
    float* __restrict__ e_out)
{
    __shared__ _Float16 sWall[WEDGE_H];      // W0 | W1 | We packed, 88 KB
    __shared__ float    sStackF[4][16 * DIN];// 48 KB (per-wave f32 staging)
    __shared__ _Float16 sH[4][16 * S1];      // 16 KB (per-wave h staging)
    __shared__ int      sSrc[4][16], sDst[4][16];
    _Float16* sW0 = sWall;
    _Float16* sW1 = sWall + W0_H;
    _Float16* sWe = sWall + W0_H + W1_H;

#ifdef HAVE_TDM
    if (threadIdx.x < 32) {
        // 90112 B = 2048 dwords/row x 11 rows
        tdm_load_2d((unsigned int)(uintptr_t)sWall, packW, 2048u, 11u);
    }
#else
    for (int i = threadIdx.x; i < WEDGE_H / 8; i += 128)
        ((int4*)sWall)[i] = ((const int4*)packW)[i];
#endif
    __syncthreads();

    const int lane = threadIdx.x & 31;
    const int w    = threadIdx.x >> 5;
    const int e    = lane & 15;        // edge row within tile
    const int p    = lane >> 4;        // which half of the wave
    const int col  = lane & 15;        // output column within n-tile
    float*    stack = sStackF[w];
    _Float16* H     = sH[w];

    // hoist loop-invariant biases / slopes into registers
    float b0v[8], b1v[8], slv[8], bev[2];
    #pragma unroll
    for (int nt = 0; nt < 8; nt++) {
        b0v[nt] = b0[nt * 16 + col];
        b1v[nt] = b1[nt * 16 + col];
        slv[nt] = prelu_slope[nt * 16 + col];
    }
    bev[0] = be[col]; bev[1] = be[16 + col];

    const int nTiles = NE / 16;
    const int waveStride = gridDim.x * 4;

    for (int tile = blockIdx.x * 4 + w; tile < nTiles; tile += waveStride) {
        const int base = tile * 16;
        const int edge = base + e;
        const int s = a_in[2 * edge], d = a_in[2 * edge + 1];
        const int rev = (edge >= E_HALF) ? (edge - E_HALF) : (edge + E_HALF);
        if (p == 0) { sSrc[w][e] = s; sDst[w][e] = d; }

        int nextTile = tile + waveStride;
        if (nextTile < nTiles) {
            __builtin_prefetch(a_in + 2 * (nextTile * 16 + e), 0, 1);
            __builtin_prefetch(e_in + (size_t)(nextTile * 16 + e) * SIN, 0, 1);
        }

        // ---- async gather: stack = [x[src] | x[dst] | e | e[rev]] (f32 rows) ----
        {
            const float* xs = x_in + (size_t)s * FIN + p * 32;
            float* o0 = stack + e * DIN + p * 32;
            #pragma unroll
            for (int j = 0; j < 8; j++) copy16ToLds(o0 + 4 * j, xs + 4 * j);
            const float* xd = x_in + (size_t)d * FIN + p * 32;
            float* o1 = stack + e * DIN + 64 + p * 32;
            #pragma unroll
            for (int j = 0; j < 8; j++) copy16ToLds(o1 + 4 * j, xd + 4 * j);
            const float* ef = e_in + (size_t)edge * SIN + p * 16;
            float* o2 = stack + e * DIN + 128 + p * 16;
            #pragma unroll
            for (int j = 0; j < 4; j++) copy16ToLds(o2 + 4 * j, ef + 4 * j);
            const float* er = e_in + (size_t)rev * SIN + p * 16;
            float* o3 = stack + e * DIN + 160 + p * 16;
            #pragma unroll
            for (int j = 0; j < 4; j++) copy16ToLds(o3 + 4 * j, er + 4 * j);
        }
        stageWait();                       // drain ASYNCcnt
        __builtin_amdgcn_wave_barrier();   // same-wave DS ops are in-order

        // ---- GEMM1: [16 x 192] @ W0 [192 x 128], ReLU ----
        v16h A1[6];
        #pragma unroll
        for (int kt = 0; kt < 6; kt++) A1[kt] = ldsA32(stack, DIN, kt, lane);
        {
            v16h Bc[6], Bn[6];
            #pragma unroll
            for (int kt = 0; kt < 6; kt++) Bc[kt] = ldsB(sW0 + kt * 512, lane);
            #pragma unroll
            for (int nt = 0; nt < 8; nt++) {
                #pragma unroll
                for (int kt = 0; kt < 6; kt++)
                    Bn[kt] = ldsB(sW0 + (((nt + 1) & 7) * 6 + kt) * 512, lane);
                v8f c;
                #pragma unroll
                for (int v = 0; v < 8; v++) c[v] = b0v[nt];
                #pragma unroll
                for (int kt = 0; kt < 6; kt++) c = wmma_f16(A1[kt], Bc[kt], c);
                #pragma unroll
                for (int v = 0; v < 8; v++) {
                    float hv = c[v] > 0.f ? c[v] : 0.f;
                    H[(v + p * 8) * S1 + nt * 16 + col] = (_Float16)hv;
                }
                #pragma unroll
                for (int kt = 0; kt < 6; kt++) Bc[kt] = Bn[kt];
            }
        }
        __builtin_amdgcn_wave_barrier();

        // ---- GEMM2: [16 x 128] @ W1 [128 x 128], PReLU, scatter-add, store h ----
        v16h A2[4];
        #pragma unroll
        for (int kt = 0; kt < 4; kt++) A2[kt] = ldsA(H, S1, kt, lane);
        {
            v16h Bc[4], Bn[4];
            #pragma unroll
            for (int kt = 0; kt < 4; kt++) Bc[kt] = ldsB(sW1 + kt * 512, lane);
            #pragma unroll
            for (int nt = 0; nt < 8; nt++) {
                #pragma unroll
                for (int kt = 0; kt < 4; kt++)
                    Bn[kt] = ldsB(sW1 + (((nt + 1) & 7) * 4 + kt) * 512, lane);
                v8f c;
                #pragma unroll
                for (int v = 0; v < 8; v++) c[v] = b1v[nt];
                #pragma unroll
                for (int kt = 0; kt < 4; kt++) c = wmma_f16(A2[kt], Bc[kt], c);
                #pragma unroll
                for (int v = 0; v < 8; v++) {
                    float hv = c[v];
                    hv = hv > 0.f ? hv : slv[nt] * hv;
                    int m = v + p * 8;
                    H[m * S1 + nt * 16 + col] = (_Float16)hv;   // A2 already in regs
                    int sm = sSrc[w][m], dm = sDst[w][m];
                    atomicAdd(outgoing + (size_t)sm * S1 + nt * 16 + col, hv);
                    atomicAdd(incoming + (size_t)dm * S1 + nt * 16 + col, hv);
                }
                #pragma unroll
                for (int kt = 0; kt < 4; kt++) Bc[kt] = Bn[kt];
            }
        }
        __builtin_amdgcn_wave_barrier();

        // ---- GEMM3: [16 x 128] @ We [128 x 32], ReLU -> e_out ----
        v16h A3[4];
        #pragma unroll
        for (int kt = 0; kt < 4; kt++) A3[kt] = ldsA(H, S1, kt, lane);
        {
            v16h Bc[4], Bn[4];
            #pragma unroll
            for (int kt = 0; kt < 4; kt++) Bc[kt] = ldsB(sWe + kt * 512, lane);
            #pragma unroll
            for (int nt = 0; nt < 2; nt++) {
                #pragma unroll
                for (int kt = 0; kt < 4; kt++)
                    Bn[kt] = ldsB(sWe + (((nt + 1) & 1) * 4 + kt) * 512, lane);
                v8f c;
                #pragma unroll
                for (int v = 0; v < 8; v++) c[v] = bev[nt];
                #pragma unroll
                for (int kt = 0; kt < 4; kt++) c = wmma_f16(A3[kt], Bc[kt], c);
                #pragma unroll
                for (int v = 0; v < 8; v++) {
                    int m = v + p * 8;
                    float r = c[v] > 0.f ? c[v] : 0.f;
                    e_out[(size_t)(base + m) * SOUT + nt * 16 + col] = r;
                }
                #pragma unroll
                for (int kt = 0; kt < 4; kt++) Bc[kt] = Bn[kt];
            }
        }
        __builtin_amdgcn_wave_barrier();
    }
}

// ---------------- kernel 2: node MLP [32768 x 320] @ Wx [320 x 64], ReLU ----------------
__global__ __launch_bounds__(128) void node_kernel(
    const float* __restrict__ x_in, const float* __restrict__ incoming,
    const float* __restrict__ outgoing, const float* __restrict__ bx,
    const _Float16* __restrict__ packWx, float* __restrict__ x_out)
{
    __shared__ _Float16 sWx[WX_H];          // 40 KB
    __shared__ float    sXF[4][16 * 320];   // 80 KB (f32 staging)

#ifdef HAVE_TDM
    if (threadIdx.x < 32) {
        // 40960 B = 2048 dwords/row x 5 rows
        tdm_load_2d((unsigned int)(uintptr_t)sWx, packWx, 2048u, 5u);
    }
#else
    for (int i = threadIdx.x; i < WX_H / 8; i += 128)
        ((int4*)sWx)[i] = ((const int4*)packWx)[i];
#endif
    __syncthreads();

    const int lane = threadIdx.x & 31;
    const int w    = threadIdx.x >> 5;
    const int r    = lane & 15;
    const int p    = lane >> 4;
    const int col  = lane & 15;
    float* X = sXF[w];

    float bxv[4];
    #pragma unroll
    for (int nt = 0; nt < 4; nt++) bxv[nt] = bx[nt * 16 + col];

    const int nTiles = N_NODES / 16;
    for (int tile = blockIdx.x * 4 + w; tile < nTiles; tile += gridDim.x * 4) {
        const int node = tile * 16 + r;
        const float* xs = x_in + (size_t)node * FIN + p * 32;
        float* o0 = X + r * 320 + p * 32;
        #pragma unroll
        for (int j = 0; j < 8; j++) copy16ToLds(o0 + 4 * j, xs + 4 * j);
        const float* ic = incoming + (size_t)node * S1 + p * 64;
        float* o1 = X + r * 320 + 64 + p * 64;
        #pragma unroll
        for (int j = 0; j < 16; j++) copy16ToLds(o1 + 4 * j, ic + 4 * j);
        const float* og = outgoing + (size_t)node * S1 + p * 64;
        float* o2 = X + r * 320 + 192 + p * 64;
        #pragma unroll
        for (int j = 0; j < 16; j++) copy16ToLds(o2 + 4 * j, og + 4 * j);
        stageWait();
        __builtin_amdgcn_wave_barrier();

        v16h A[10];
        #pragma unroll
        for (int kt = 0; kt < 10; kt++) A[kt] = ldsA32(X, 320, kt, lane);
        v16h Bc[10], Bn[10];
        #pragma unroll
        for (int kt = 0; kt < 10; kt++) Bc[kt] = ldsB(sWx + kt * 512, lane);
        #pragma unroll
        for (int nt = 0; nt < 4; nt++) {
            #pragma unroll
            for (int kt = 0; kt < 10; kt++)
                Bn[kt] = ldsB(sWx + (((nt + 1) & 3) * 10 + kt) * 512, lane);
            v8f c;
            #pragma unroll
            for (int v = 0; v < 8; v++) c[v] = bxv[nt];
            #pragma unroll
            for (int kt = 0; kt < 10; kt++) c = wmma_f16(A[kt], Bc[kt], c);
            #pragma unroll
            for (int v = 0; v < 8; v++) {
                int m = v + p * 8;
                float rr = c[v] > 0.f ? c[v] : 0.f;
                x_out[(size_t)(tile * 16 + m) * FOUT + nt * 16 + col] = rr;
            }
            #pragma unroll
            for (int kt = 0; kt < 10; kt++) Bc[kt] = Bn[kt];
        }
        __builtin_amdgcn_wave_barrier();
    }
}

// ---------------- kernel 3: finalize error flag ----------------
__global__ void flag_kernel(const int* flagInt, float* out) {
    if (threadIdx.x == 0 && blockIdx.x == 0) *out = (*flagInt) ? 1.0f : 0.0f;
}

// ---------------- host glue ----------------
extern "C" void kernel_launch(void* const* d_in, const int* in_sizes, int n_in,
                              void* d_out, int out_size, void* d_ws, size_t ws_size,
                              hipStream_t stream) {
    const float* x_in = (const float*)d_in[0];
    const int*   a_in = (const int*)d_in[1];
    const float* e_in = (const float*)d_in[2];
    const float* W0   = (const float*)d_in[3];
    const float* b0   = (const float*)d_in[4];
    const float* W1   = (const float*)d_in[5];
    const float* b1   = (const float*)d_in[6];
    const float* prelu= (const float*)d_in[7];
    const float* Wx   = (const float*)d_in[8];
    const float* bx   = (const float*)d_in[9];
    const float* We   = (const float*)d_in[10];
    const float* be   = (const float*)d_in[11];

    float* x_out   = (float*)d_out;
    float* e_out   = x_out + (size_t)N_NODES * FOUT;
    float* flagOut = e_out + (size_t)NE * SOUT;

    char* ws = (char*)d_ws;
    float* incoming = (float*)ws;                                   // 16 MB
    float* outgoing = incoming + (size_t)N_NODES * S1;              // 16 MB
    int*   flagInt  = (int*)(ws + (size_t)2 * N_NODES * S1 * 4);
    _Float16* packW  = (_Float16*)(ws + (size_t)2 * N_NODES * S1 * 4 + 64);
    _Float16* packWx = packW + WEDGE_H;

    const int accum4 = (2 * N_NODES * S1) / 4;
    zero_kernel<<<2048, 256, 0, stream>>>((float4*)incoming, accum4, flagInt);
    pack_kernel<<<(PACK_TOTAL_H + 255) / 256, 256, 0, stream>>>(W0, W1, We, Wx, packW);
    check_kernel<<<NE / 256, 256, 0, stream>>>(a_in, flagInt);
    edge_kernel<<<1024, 128, 0, stream>>>(x_in, a_in, e_in, b0, b1, prelu, be,
                                          packW, incoming, outgoing, e_out);
    node_kernel<<<512, 128, 0, stream>>>(x_in, incoming, outgoing, bx, packWx, x_out);
    flag_kernel<<<1, 32, 0, stream>>>(flagInt, flagOut);
}
